// Hallucigraph_32409823216072
// MI455X (gfx1250) — compile-verified, matrix operand order checked by reading, not run on hardware
//
#include <hip/hip_runtime.h>
#include <cstdint>
#include <cstddef>

// ---------------------------------------------------------------------------
// Problem constants (from reference setup_inputs)
// ---------------------------------------------------------------------------
#define NN_   6144
#define FF_   512
#define HH_   256
#define CC_   40
#define TEMP_INV 4.0f     // 1/0.25
#define ALPHA_   0.5f
#define EPS_     1e-10f

typedef __attribute__((ext_vector_type(16))) __bf16 bf16x16;
typedef __attribute__((ext_vector_type(8)))  float  floatx8;

struct __attribute__((aligned(8))) u16x4 { uint16_t x, y, z, w; };

// ---------------------------------------------------------------------------
// Helpers
// ---------------------------------------------------------------------------
__device__ __forceinline__ uint16_t f2bf(float f) {
    uint32_t u = __float_as_uint(f);
    uint32_t r = u + 0x7FFFu + ((u >> 16) & 1u);   // round-to-nearest-even
    return (uint16_t)(r >> 16);
}

__device__ __forceinline__ uint32_t rotl32(uint32_t x, int r) {
    return (x << r) | (x >> (32 - r));
}

// threefry2x32 with key = [0, 42] (jax.random.key(42)); JAX iota-split
// counter mapping: elem i < total/2 -> x0 of ctr (i, i+half); else x1.
__device__ float jax_uniform(size_t idx, size_t total) {
    const uint32_t ks0 = 0u, ks1 = 42u;
    const uint32_t ks2 = 0x1BD11BDAu ^ ks0 ^ ks1;
    uint32_t half = (uint32_t)(total >> 1);
    uint32_t c0, c1; int which;
    if (idx < (size_t)half) { c0 = (uint32_t)idx;          c1 = (uint32_t)idx + half; which = 0; }
    else                    { c0 = (uint32_t)(idx - half); c1 = (uint32_t)idx;        which = 1; }
    uint32_t x0 = c0 + ks0, x1 = c1 + ks1;
    const int RA[4] = {13, 15, 26, 6};
    const int RB[4] = {17, 29, 16, 24};
#define TF_ROUND(r) { x0 += x1; x1 = rotl32(x1, (r)); x1 ^= x0; }
    for (int i = 0; i < 4; ++i) TF_ROUND(RA[i]);  x0 += ks1; x1 += ks2 + 1u;
    for (int i = 0; i < 4; ++i) TF_ROUND(RB[i]);  x0 += ks2; x1 += ks0 + 2u;
    for (int i = 0; i < 4; ++i) TF_ROUND(RA[i]);  x0 += ks0; x1 += ks1 + 3u;
    for (int i = 0; i < 4; ++i) TF_ROUND(RB[i]);  x0 += ks1; x1 += ks2 + 4u;
    for (int i = 0; i < 4; ++i) TF_ROUND(RA[i]);  x0 += ks2; x1 += ks0 + 5u;
#undef TF_ROUND
    uint32_t bits = which ? x1 : x0;
    float f01 = __uint_as_float((bits >> 9) | 0x3F800000u) - 1.0f;   // [0,1)
    return EPS_ + (1.0f - 2.0f * EPS_) * f01;                         // [eps,1-eps)
}

// Binary-concrete relaxation sample, recomputed from the raw score.
__device__ float sample_rec(float s, size_t idx, size_t total) {
    float p      = 1.0f / (1.0f + __expf(-s));
    float logits = __logf(p + EPS_) - __logf(1.0f - p + EPS_);
    float u      = jax_uniform(idx, total);
    float noise  = __logf(u) - log1pf(-u);
    return 1.0f / (1.0f + __expf(-(logits + noise) * TEMP_INV));
}

// ---------------------------------------------------------------------------
// fp32 -> bf16 conversion (vectorized; n must be a multiple of 4 — true for
// every buffer in this pipeline)
// ---------------------------------------------------------------------------
__global__ __launch_bounds__(256) void k_f32_to_bf16(const float* __restrict__ in,
                                                     uint16_t* __restrict__ out, size_t n4) {
    size_t i = (size_t)blockIdx.x * 256 + threadIdx.x;
    if (i >= n4) return;
    float4 v = ((const float4*)in)[i];
    u16x4 o;
    o.x = f2bf(v.x); o.y = f2bf(v.y); o.z = f2bf(v.z); o.w = f2bf(v.w);
    ((u16x4*)out)[i] = o;
}

// fp32 [R][Cc] -> bf16 transposed [Cc][R], LDS-tiled 32x32
__global__ __launch_bounds__(256) void k_f32_to_bf16_T(const float* __restrict__ in,
                                                       uint16_t* __restrict__ out,
                                                       int R, int Cc) {
    __shared__ uint16_t tile[32][34];
    int tx = threadIdx.x & 31;
    int ty = threadIdx.x >> 5;              // 0..7
    int r0 = blockIdx.y * 32;
    int c0 = blockIdx.x * 32;
    for (int j = 0; j < 4; ++j) {
        int r = r0 + ty + j * 8;
        int c = c0 + tx;
        uint16_t v = 0;
        if (r < R && c < Cc) v = f2bf(in[(size_t)r * Cc + c]);
        tile[ty + j * 8][tx] = v;
    }
    __syncthreads();
    for (int j = 0; j < 4; ++j) {
        int c = c0 + ty + j * 8;            // output row = original column
        int r = r0 + tx;                    // output col = original row
        if (c < Cc && r < R) out[(size_t)c * R + r] = tile[tx][ty + j * 8];
    }
}

// out_bf = bf16(a + b)
__global__ __launch_bounds__(256) void k_sum_to_bf16(const float* __restrict__ a,
                                                     const float* __restrict__ b,
                                                     uint16_t* __restrict__ out, size_t n) {
    size_t i = (size_t)blockIdx.x * 256 + threadIdx.x;
    if (i < n) out[i] = f2bf(a[i] + b[i]);
}

// out_bf = bf16((1-alpha)*a + alpha*b)
__global__ __launch_bounds__(256) void k_blend_to_bf16(const float* __restrict__ a,
                                                       const float* __restrict__ b,
                                                       uint16_t* __restrict__ out, size_t n) {
    size_t i = (size_t)blockIdx.x * 256 + threadIdx.x;
    if (i < n) out[i] = f2bf((1.0f - ALPHA_) * a[i] + ALPHA_ * b[i]);
}

// rs[i] = rsqrt(1 + sum_j sample_rec(scores[i,j]))     (row of A+I)
__global__ __launch_bounds__(256) void k_rowsum_rsqrt(const float* __restrict__ scores,
                                                      float* __restrict__ rs, int n) {
    __shared__ float red[256];
    size_t total = (size_t)n * n;
    int i = blockIdx.x;
    float s = 0.0f;
    for (int j = threadIdx.x; j < n; j += 256) {
        size_t idx = (size_t)i * n + j;
        s += sample_rec(scores[idx], idx, total);
    }
    red[threadIdx.x] = s;
    __syncthreads();
    for (int st = 128; st > 0; st >>= 1) {
        if ((int)threadIdx.x < st) red[threadIdx.x] += red[threadIdx.x + st];
        __syncthreads();
    }
    if (threadIdx.x == 0) rs[i] = rsqrtf(red[0] + 1.0f);
}

// adjn_bf[i,j] = bf16((sample_rec + (i==j)) * rs[i] * rs[j])
__global__ __launch_bounds__(256) void k_normalize_bf16(const float* __restrict__ scores,
                                                        const float* __restrict__ rs,
                                                        uint16_t* __restrict__ out, int n) {
    size_t total = (size_t)n * n;
    size_t idx = (size_t)blockIdx.x * 256 + threadIdx.x;
    if (idx >= total) return;
    int i = (int)(idx / n);
    int j = (int)(idx % n);
    float a = sample_rec(scores[idx], idx, total) + (i == j ? 1.0f : 0.0f);
    out[idx] = f2bf(a * rs[i] * rs[j]);
}

// per-row log_softmax over C=40 columns; 64 threads / row
__global__ __launch_bounds__(64) void k_log_softmax(const float* __restrict__ preds,
                                                    float* __restrict__ out, int C) {
    __shared__ float red[64];
    int row = blockIdx.x;
    int t = threadIdx.x;
    float v = (t < C) ? preds[(size_t)row * C + t] : -1e30f;
    red[t] = v;
    __syncthreads();
    for (int s = 32; s > 0; s >>= 1) {
        if (t < s) red[t] = fmaxf(red[t], red[t + s]);
        __syncthreads();
    }
    float m = red[0];
    __syncthreads();
    float e = (t < C) ? __expf(v - m) : 0.0f;
    red[t] = e;
    __syncthreads();
    for (int s = 32; s > 0; s >>= 1) {
        if (t < s) red[t] += red[t + s];
        __syncthreads();
    }
    float lse = m + __logf(red[0]);
    if (t < C) out[(size_t)row * C + t] = v - lse;
}

// ---------------------------------------------------------------------------
// bf16 WMMA GEMM: C[M,N] = A[M,K] @ Bt[N,K]^T   (f32 accumulate)
//   Every B operand is pre-transposed to [N][K] so both tiles stage as
//   contiguous [row][K] runs.
//   BOUNDS = 1: guard the N dimension (only the C=40 GEMMs need it)
//   ACT    = 1: fused ReLU epilogue
// Tile: BM=BN=128, BK=32. 256 threads = 8 waves (4x2); each wave computes a
// 32x64 sub-tile = 2x4 v_wmma_f32_16x16x32_bf16 accumulators.
// Staging: global_load_async_to_lds_b128 (ASYNCcnt) into double-buffered LDS,
// so tile t+1 DMA overlaps tile t fragment ds_loads + WMMAs.
// LDS row stride 40 elems (80 B) keeps b128 targets 16-byte aligned.
// ---------------------------------------------------------------------------
#define BM 128
#define BN 128
#define BK 32
#define LDK 40

__device__ __forceinline__ void async_b128(uint16_t* lds_dst, const uint16_t* gsrc) {
    uint32_t laddr = (uint32_t)(uintptr_t)lds_dst;   // generic->LDS offset (addr[31:0])
    asm volatile("global_load_async_to_lds_b128 %0, %1, off"
                 :: "v"(laddr), "v"(gsrc) : "memory");
}
__device__ __forceinline__ void wait_async0() {
    asm volatile("s_wait_asynccnt 0x0" ::: "memory");
}

template <int BOUNDS, int ACT>
__global__ __launch_bounds__(256) void k_gemm_bf16t(const uint16_t* __restrict__ A,
                                                    const uint16_t* __restrict__ Bt,
                                                    float* __restrict__ C,
                                                    int M, int N, int K) {
    __shared__ uint16_t sh[2][2][BM * LDK];          // [buf][A/B][tile]

    const int tid  = threadIdx.x;
    const int lane = tid & 31;
    const int wave = tid >> 5;
    const int half = lane >> 4;
    const int l16  = lane & 15;
    const int wm   = wave >> 1;          // 0..3 -> 32-row group
    const int wn   = wave & 1;           // 0..1 -> 64-col group
    const int m0   = blockIdx.y * BM;
    const int n0   = blockIdx.x * BN;

    floatx8 acc[2][4];
    for (int i = 0; i < 2; ++i)
        for (int j = 0; j < 4; ++j)
            for (int v = 0; v < 8; ++v) acc[i][j][v] = 0.0f;

    // Stage one BKxBM A-tile and BKxBN B-tile into LDS buffer `buf` (async).
    auto stage = [&](int buf, int k0) {
        #pragma unroll
        for (int r = 0; r < 2; ++r) {
            int id = tid + r * 256;      // 0..511
            int m  = id >> 2;            // 0..127
            int c  = id & 3;             // 16-byte chunk (8 bf16)
            async_b128(&sh[buf][0][m * LDK + c * 8],
                       A + (size_t)(m0 + m) * K + (k0 + c * 8));
        }
        #pragma unroll
        for (int r = 0; r < 2; ++r) {
            int id = tid + r * 256;
            int n  = id >> 2;
            int c  = id & 3;
            uint16_t* l = &sh[buf][1][n * LDK + c * 8];
            if (!BOUNDS || (n0 + n) < N) {
                async_b128(l, Bt + (size_t)(n0 + n) * K + (k0 + c * 8));
            } else {
                uint32_t* z = (uint32_t*)l;
                z[0] = 0u; z[1] = 0u; z[2] = 0u; z[3] = 0u;
            }
        }
    };

    stage(0, 0);
    wait_async0();
    __syncthreads();

    const int nk = K / BK;               // K is always a multiple of 32 here
    for (int t = 0; t < nk; ++t) {
        const int buf = t & 1;
        if (t + 1 < nk) stage(buf ^ 1, (t + 1) * BK);

        const uint16_t* la = sh[buf][0];
        const uint16_t* lb = sh[buf][1];

        // Fragments (16-bit A-matrix layout, ISA 7.12.2)
        bf16x16 afrag[2], bfrag[4];
        for (int i = 0; i < 2; ++i) {
            int row = wm * 32 + i * 16 + l16;
            union { uint32_t u[8]; bf16x16 v; } f;
            for (int v = 0; v < 8; ++v) {
                int kb = ((v & 4) ? 16 : 0) + half * 8 + (v & 3) * 2;
                f.u[v] = *(const uint32_t*)(la + row * LDK + kb);
            }
            afrag[i] = f.v;
        }
        for (int j = 0; j < 4; ++j) {
            int col = wn * 64 + j * 16 + l16;
            union { uint32_t u[8]; bf16x16 v; } f;
            for (int v = 0; v < 8; ++v) {
                int kb = ((v & 4) ? 16 : 0) + half * 8 + (v & 3) * 2;
                f.u[v] = *(const uint32_t*)(lb + col * LDK + kb);
            }
            bfrag[j] = f.v;
        }

        for (int i = 0; i < 2; ++i)
            for (int j = 0; j < 4; ++j)
                acc[i][j] = __builtin_amdgcn_wmma_f32_16x16x32_bf16(
                    false, afrag[i], false, bfrag[j],
                    (short)0, acc[i][j], false, false);

        wait_async0();                   // next tile's DMA done
        __syncthreads();                 // everyone finished reading `buf`
    }

    // Epilogue (C/D layout: VGPR v -> local row v + 8*half; col = l16)
    for (int i = 0; i < 2; ++i)
        for (int j = 0; j < 4; ++j)
            for (int v = 0; v < 8; ++v) {
                int row = m0 + wm * 32 + i * 16 + v + half * 8;
                int col = n0 + wn * 64 + j * 16 + l16;
                float x = acc[i][j][v];
                if (ACT) x = fmaxf(x, 0.0f);
                if (!BOUNDS || col < N)
                    C[(size_t)row * N + col] = x;
            }
}

// ---------------------------------------------------------------------------
// Host-side pipeline
// ---------------------------------------------------------------------------
static inline dim3 gemm_grid(int M, int N) {
    return dim3((N + BN - 1) / BN, (M + BM - 1) / BM, 1);
}
static inline dim3 tr_grid(int R, int Cc) {
    return dim3((Cc + 31) / 32, (R + 31) / 32, 1);
}

extern "C" void kernel_launch(void* const* d_in, const int* in_sizes, int n_in,
                              void* d_out, int out_size, void* d_ws, size_t ws_size,
                              hipStream_t stream) {
    (void)in_sizes; (void)n_in; (void)out_size; (void)ws_size;

    const int N = NN_, F = FF_, H = HH_, C = CC_;
    const size_t NNe = (size_t)N * N;

    const float* x   = (const float*)d_in[0];
    const float* adj = (const float*)d_in[1];
    const float* W1  = (const float*)d_in[2];
    const float* W2  = (const float*)d_in[3];
    const float* W3  = (const float*)d_in[4];
    const float* Wd1 = (const float*)d_in[5];
    const float* Wd2 = (const float*)d_in[6];
    const float* Wd3 = (const float*)d_in[7];

    float* out_scores = (float*)d_out;            // [N,N]
    float* out_lsm    = out_scores + NNe;         // [N,C]

    // ---- carve workspace (~185 MB total)
    char* p = (char*)d_ws;
    auto alloc = [&](size_t bytes) -> void* {
        void* r = (void*)p;
        p += (bytes + 255) & ~(size_t)255;
        return r;
    };
    uint16_t* adj_bf  = (uint16_t*)alloc(NNe * 2);
    uint16_t* adjn_bf = (uint16_t*)alloc(NNe * 2);
    uint16_t* x_bf    = (uint16_t*)alloc((size_t)N * F * 2);
    uint16_t* w1t     = (uint16_t*)alloc((size_t)F * H * 2);   // [H][F]
    uint16_t* w2t     = (uint16_t*)alloc((size_t)H * H * 2);   // [H][H]
    uint16_t* w3t     = (uint16_t*)alloc((size_t)H * C * 2);   // [C][H]
    uint16_t* wd1t    = (uint16_t*)alloc((size_t)H * H * 2);
    uint16_t* wd2t    = (uint16_t*)alloc((size_t)F * H * 2);   // [H][F]
    uint16_t* wd3t    = (uint16_t*)alloc((size_t)H * H * 2);
    float*    bufA    = (float*)alloc((size_t)N * H * 4);      // pre-products / S6
    float*    bufB    = (float*)alloc((size_t)N * H * 4);      // h, z1
    float*    bufC    = (float*)alloc((size_t)N * H * 4);      // z0
    float*    bufD    = (float*)alloc((size_t)N * H * 4);      // z2, z3
    uint16_t* tA      = (uint16_t*)alloc((size_t)N * H * 2);   // row-major bf16 temps
    uint16_t* tB      = (uint16_t*)alloc((size_t)N * H * 2);   // transposed bf16 temps
    float*    rs      = (float*)alloc((size_t)N * 4);
    float*    preds   = (float*)alloc((size_t)N * C * 4);

    auto cvt = [&](const float* src, uint16_t* dst, size_t n) {
        size_t n4 = n / 4;
        k_f32_to_bf16<<<(unsigned)((n4 + 255) / 256), 256, 0, stream>>>(src, dst, n4);
    };
    auto cvtT = [&](const float* src, uint16_t* dst, int R, int Cc) {
        k_f32_to_bf16_T<<<tr_grid(R, Cc), 256, 0, stream>>>(src, dst, R, Cc);
    };

    // ---- bf16 copies: A-operands row-major, B-operands transposed
    cvt(adj, adj_bf, NNe);
    cvt(x, x_bf, (size_t)N * F);
    cvtT(W1, w1t, F, H);
    cvtT(W2, w2t, H, H);
    cvtT(W3, w3t, H, C);
    cvtT(Wd1, wd1t, H, H);
    cvtT(Wd2, wd2t, F, H);
    cvtT(Wd3, wd3t, H, H);

    // ---- encoder
    // S1 = x @ W1 ; h = relu(adj @ S1)
    k_gemm_bf16t<0, 0><<<gemm_grid(N, H), 256, 0, stream>>>(x_bf, w1t, bufA, N, H, F);
    cvtT(bufA, tB, N, H);
    k_gemm_bf16t<0, 1><<<gemm_grid(N, H), 256, 0, stream>>>(adj_bf, tB, bufB, N, H, N);
    cvt(bufB, tA, (size_t)N * H);
    // S2 = h @ W2 ; z0 = adj @ S2
    k_gemm_bf16t<0, 0><<<gemm_grid(N, H), 256, 0, stream>>>(tA, w2t, bufA, N, H, H);
    cvtT(bufA, tB, N, H);
    k_gemm_bf16t<0, 0><<<gemm_grid(N, H), 256, 0, stream>>>(adj_bf, tB, bufC, N, H, N);
    cvt(bufC, tA, (size_t)N * H);       // tA = z0_bf [N][H] (A-layout == Bt-layout here)

    // ---- adjacency reconstruction
    // scores = z0 @ z0^T (first output)
    k_gemm_bf16t<0, 0><<<gemm_grid(N, N), 256, 0, stream>>>(tA, tA, out_scores, N, N, H);
    k_rowsum_rsqrt<<<N, 256, 0, stream>>>(out_scores, rs, N);
    k_normalize_bf16<<<(unsigned)((NNe + 255) / 256), 256, 0, stream>>>(out_scores, rs, adjn_bf, N);

    // ---- decoder
    // S3 = z0 @ Wd1 ; z1 = relu(adj_norm @ S3)
    k_gemm_bf16t<0, 0><<<gemm_grid(N, H), 256, 0, stream>>>(tA, wd1t, bufA, N, H, H);
    cvtT(bufA, tB, N, H);
    k_gemm_bf16t<0, 1><<<gemm_grid(N, H), 256, 0, stream>>>(adjn_bf, tB, bufB, N, H, N);
    // S4 = x @ Wd2 ; z2 = relu(adj_norm @ S4)
    k_gemm_bf16t<0, 0><<<gemm_grid(N, H), 256, 0, stream>>>(x_bf, wd2t, bufA, N, H, F);
    cvtT(bufA, tB, N, H);
    k_gemm_bf16t<0, 1><<<gemm_grid(N, H), 256, 0, stream>>>(adjn_bf, tB, bufD, N, H, N);
    // (z1+z2) ; S5 = (z1+z2) @ Wd3 ; z3 = adj_norm @ S5
    k_sum_to_bf16<<<(unsigned)(((size_t)N * H + 255) / 256), 256, 0, stream>>>(bufB, bufD, tA, (size_t)N * H);
    k_gemm_bf16t<0, 0><<<gemm_grid(N, H), 256, 0, stream>>>(tA, wd3t, bufA, N, H, H);
    cvtT(bufA, tB, N, H);
    k_gemm_bf16t<0, 0><<<gemm_grid(N, H), 256, 0, stream>>>(adjn_bf, tB, bufD, N, H, N);  // z3

    // ---- z = (1-a)*z0 + a*z3 ; preds = adj @ (z @ W3) ; log_softmax
    k_blend_to_bf16<<<(unsigned)(((size_t)N * H + 255) / 256), 256, 0, stream>>>(bufC, bufD, tA, (size_t)N * H);
    k_gemm_bf16t<1, 0><<<gemm_grid(N, C), 256, 0, stream>>>(tA, w3t, bufA, N, C, H);      // S6 [N,40]
    cvtT(bufA, tB, N, C);                                                                 // [40][6144]
    k_gemm_bf16t<1, 0><<<gemm_grid(N, C), 256, 0, stream>>>(adj_bf, tB, preds, N, C, N);
    k_log_softmax<<<N, 64, 0, stream>>>(preds, out_lsm, C);
}
